// _UserModel_32469952757837
// MI455X (gfx1250) — matrix operand order repaired and checked by reading, not compiled, not run on hardware
//
#include <hip/hip_runtime.h>

#define B_  512
#define P_  50
#define Q_  30
#define P2_ 30
#define D_  64

// padded LDS leading dims (bf16 elements): 136*2B=272B=68 dwords, 72*2B=144B=36 dwords
// -> row-bank stride mod 64 is 4 / 36: the 16 rows of a half-wave map to 16 distinct
//    4-bank groups => conflict-free ds_load_b128 fragment loads.
#define LD128 136
#define LD64  72

typedef __attribute__((ext_vector_type(16))) __bf16 v16bf;
typedef __attribute__((ext_vector_type(8)))  __bf16 v8bf;
typedef __attribute__((ext_vector_type(8)))  float  v8f;

__device__ __forceinline__ void st4bf(__bf16* dst, float4 v) {
    union { __bf16 h[4]; uint2 u; } p;
    p.h[0] = (__bf16)v.x; p.h[1] = (__bf16)v.y; p.h[2] = (__bf16)v.z; p.h[3] = (__bf16)v.w;
    *(uint2*)dst = p.u;
}

// copy (rows x cols) f32 row-major -> LDS bf16 with leading dim ld (cols % 4 == 0)
__device__ __forceinline__ void cvt_cp2d(const float* g, __bf16* s, int rows, int cols, int ld,
                                         int tid, int nt) {
    const int q = cols >> 2, tot = rows * q;
    for (int i = tid; i < tot; i += nt) {
        const int r = i / q, c4 = (i - r * q) << 2;
        st4bf(s + r * ld + c4, *(const float4*)(g + r * cols + c4));
    }
}
__device__ __forceinline__ void cp_f(const float* g, float* s, int n, int tid, int nt) {
    for (int i = tid; i < n; i += nt) s[i] = g[i];
}

// C(M x N) = act(A(M x K) @ W(N x K)^T + bias); A/W bf16 LDS, C bf16 LDS.
// WMMA fragment mapping per CDNA5 ISA 7.12.2:
//   A: lane L holds row M=L&15; lanes<16 K = kc+[0..7],[16..23]; lanes>=16 K = kc+[8..15],[24..31]
//   B: lane L holds col N=L&15; same K order as A
//   D: VGPR r -> row mt*16 + r (+8 for lanes>=16), col = nt*16 + (L&15)
template<int ACT, int KDIM, int MT, int NT>
__device__ __forceinline__ void gemm_bf(const __bf16* __restrict__ A, int lda,
                                        const __bf16* __restrict__ W, int wld,
                                        const float*  __restrict__ bias,
                                        __bf16* __restrict__ O, int ldo, int tid) {
    const int wave = tid >> 5, lane = tid & 31;
    const int mrow = lane & 15;
    const int ahi  = (lane & 16) ? 8 : 0;    // A k-offset AND D row offset
    const int bhi  = (lane & 16) ? 16 : 0;   // unused (B k-offset is ahi)
    for (int j = wave; j < MT * NT; j += 4) {
        const int mt = j / NT, nt = j % NT;
        v8f acc = {};
#pragma unroll
        for (int kc = 0; kc < KDIM; kc += 32) {
            union { v16bf v; v8bf h[2]; } ua, ub;
            const __bf16* ap = A + (mt * 16 + mrow) * lda + kc + ahi;
            ua.h[0] = *(const v8bf*)ap;
            ua.h[1] = *(const v8bf*)(ap + 16);
            const __bf16* bp = W + (nt * 16 + mrow) * wld + kc + ahi;
            ub.h[0] = *(const v8bf*)bp;
            ub.h[1] = *(const v8bf*)(bp + 16);
            acc = __builtin_amdgcn_wmma_f32_16x16x32_bf16(false, ua.v, false, ub.v,
                                                          (short)0, acc, false, false);
        }
        const int   n  = nt * 16 + mrow;
        const int   mb = mt * 16 + ahi;
        const float bv = bias[n];
#pragma unroll
        for (int r = 0; r < 8; r++) {
            float v = acc[r] + bv;
            if (ACT == 1) v = (v > 0.0f) ? v : 0.2f * v;   // leaky_relu 0.2
            if (ACT == 2) v = fmaxf(v, 0.0f);              // relu
            O[(mb + r) * ldo + n] = (__bf16)v;
        }
    }
}

// ---------------- social branch: per (b,q) attention over P2 items ----------------
__global__ void __launch_bounds__(128)
k_social(const int* __restrict__ uup, const int* __restrict__ uuip,
         const float* __restrict__ userT, const float* __restrict__ itemT,
         const float* __restrict__ rateT,
         const float* gv_W1, const float* gv_b1, const float* gv_W2, const float* gv_b2,
         const float* w4_W, const float* w4_b,
         const float* aW1, const float* ab1, const float* aW2, const float* ab2,
         float* __restrict__ t_s) {
    extern __shared__ __align__(16) char smem[];
    __bf16* sGV1 = (__bf16*)smem;          // 64 x LD128
    __bf16* sGV2 = sGV1 + 64 * LD128;      // 64 x LD64
    __bf16* sW4  = sGV2 + 64 * LD64;       // 64 x LD64
    __bf16* sA1  = sW4  + 64 * LD64;       // 64 x LD128
    __bf16* sBA  = sA1  + 64 * LD128;      // 32 x LD128
    __bf16* sBB  = sBA  + 32 * LD128;      // 32 x LD64
    __bf16* sBC  = sBB  + 32 * LD64;       // 32 x LD64
    float*  sb1  = (float*)(sBC + 32 * LD64);
    float*  sb2  = sb1 + 64;
    float*  sab  = sb2 + 64;
    float*  sa2  = sab + 64;
    float*  sbw  = sa2 + 64;
    float*  sUw  = sbw + 64;
    float*  sAl  = sUw + 64;   // 32
    float*  sMk  = sAl + 32;   // 32
    float*  sRd  = sMk + 32;   // 4

    const int tid = threadIdx.x, b = blockIdx.x;
    cvt_cp2d(gv_W1, sGV1, 64, 128, LD128, tid, 128);
    cvt_cp2d(gv_W2, sGV2, 64, 64, LD64, tid, 128);
    cvt_cp2d(w4_W,  sW4,  64, 64, LD64, tid, 128);
    cvt_cp2d(aW1,   sA1,  64, 128, LD128, tid, 128);
    cp_f(gv_b1, sb1, 64, tid, 128);
    cp_f(gv_b2, sb2, 64, tid, 128);
    cp_f(ab1,   sab, 64, tid, 128);
    cp_f(aW2,   sa2, 64, tid, 128);
    cp_f(w4_b,  sbw, 64, tid, 128);
    if (tid == 0) sRd[1] = ab2[0];
    __syncthreads();

    for (int q = 0; q < Q_; q++) {
        const int base = (b * Q_ + q) * P2_;
        {   // gather X = [item_emb | rate_emb] -> 32 x 128 bf16 tile, mask
            const int r = tid >> 2, c0 = (tid & 3) * 32;
            int ii = 0, ri = 0; float m = 0.0f;
            if (r < P2_) {
                ii = uuip[(base + r) * 2 + 0];
                ri = uuip[(base + r) * 2 + 1];
                m  = (ii > 0) ? 1.0f : 0.0f;
            }
            const float* src = (c0 < 64) ? (itemT + (size_t)ii * 64 + c0)
                                         : (rateT + (size_t)ri * 64 + (c0 - 64));
            const float4 z = make_float4(0.f, 0.f, 0.f, 0.f);
#pragma unroll
            for (int j = 0; j < 8; j++) {
                const float4 v = (r < P2_) ? *(const float4*)(src + 4 * j) : z;
                st4bf(sBA + r * LD128 + c0 + 4 * j, v);
            }
            if ((tid & 3) == 0) sMk[r] = m;
        }
        if (q + 1 < Q_ && (tid & 3) == 0) {   // prefetch next q's gather rows into L2/WGP$
            const int r = tid >> 2;
            if (r < P2_) {
                const int nb = (b * Q_ + q + 1) * P2_;
                const int ni = uuip[(nb + r) * 2 + 0];
                const int nr = uuip[(nb + r) * 2 + 1];
                __builtin_prefetch(itemT + (size_t)ni * 64, 0, 1);
                __builtin_prefetch(rateT + (size_t)nr * 64, 0, 1);
            }
        }
        if (tid < 64) {   // uW4 = user_emb @ w4^T (no bias)
            const int uu = uup[b * Q_ + q];
            const float* up = userT + (size_t)uu * 64;
            float s = 0.0f;
            for (int k = 0; k < 64; k++) s += up[k] * (float)sW4[tid * LD64 + k];
            sUw[tid] = s;
        }
        __syncthreads();
        gemm_bf<1, 128, 2, 4>(sBA, LD128, sGV1, LD128, sb1, sBB, LD64, tid); // leaky(X@gv1^T+b1)
        __syncthreads();
        gemm_bf<0, 64, 2, 4>(sBB, LD64, sGV2, LD64, sb2, sBC, LD64, tid);    // x_ia
        __syncthreads();
        gemm_bf<0, 64, 2, 4>(sBC, LD64, sW4, LD64, sbw, sBA, LD128, tid);    // y -> left half
        {   // right half: mask * uW4 + w4_b
            const int r = tid >> 2, c0 = (tid & 3) * 16;
            for (int c = c0; c < c0 + 16; c++)
                sBA[r * LD128 + 64 + c] = (__bf16)(sMk[r] * sUw[c] + sbw[c]);
        }
        __syncthreads();
        gemm_bf<1, 128, 2, 4>(sBA, LD128, sA1, LD128, sab, sBB, LD64, tid);  // att hidden
        __syncthreads();
        if (tid < 32) {   // logits -> exp*mask
            float s = 0.0f;
            for (int k = 0; k < 64; k++) s += (float)sBB[tid * LD64 + k] * sa2[k];
            sAl[tid] = __expf(s + sRd[1]) * sMk[tid];
        }
        __syncthreads();
        if (tid == 0) { float s = 0.0f; for (int p = 0; p < 32; p++) s += sAl[p]; sRd[0] = 1.0f / (s + 1e-10f); }
        __syncthreads();
        if (tid < 64) {   // t_s = sum_p alpha_p * x_ia[p]
            float s = 0.0f;
            for (int p = 0; p < P2_; p++) s += sAl[p] * (float)sBC[p * LD64 + tid];
            t_s[(size_t)(b * Q_ + q) * 64 + tid] = s * sRd[0];
        }
        __syncthreads();
    }
}

// ---------------- item branch: per b attention over P items ----------------
__global__ void __launch_bounds__(128)
k_item(const int* __restrict__ uids, const int* __restrict__ uip,
       const float* __restrict__ userT, const float* __restrict__ itemT,
       const float* __restrict__ rateT,
       const float* gv_W1, const float* gv_b1, const float* gv_W2, const float* gv_b2,
       const float* w1_W, const float* w1_b,
       const float* aW1, const float* ab1, const float* aW2, const float* ab2,
       float* __restrict__ t_i) {
    extern __shared__ __align__(16) char smem[];
    __bf16* sGV1 = (__bf16*)smem;          // 64 x LD128
    __bf16* sGV2 = sGV1 + 64 * LD128;      // 64 x LD64
    __bf16* sW1  = sGV2 + 64 * LD64;       // 64 x LD64
    __bf16* sA1  = sW1  + 64 * LD64;       // 64 x LD128
    __bf16* sBA  = sA1  + 64 * LD128;      // 64 x LD128
    __bf16* sBB  = sBA  + 64 * LD128;      // 64 x LD64
    __bf16* sBC  = sBB  + 64 * LD64;       // 64 x LD64
    float*  sb1  = (float*)(sBC + 64 * LD64);
    float*  sb2  = sb1 + 64;
    float*  sab  = sb2 + 64;
    float*  sa2  = sab + 64;
    float*  sbw  = sa2 + 64;
    float*  sUw  = sbw + 64;
    float*  sAl  = sUw + 64;   // 64
    float*  sMk  = sAl + 64;   // 64
    float*  sRd  = sMk + 64;   // 4

    const int tid = threadIdx.x, b = blockIdx.x;
    cvt_cp2d(gv_W1, sGV1, 64, 128, LD128, tid, 128);
    cvt_cp2d(gv_W2, sGV2, 64, 64, LD64, tid, 128);
    cvt_cp2d(w1_W,  sW1,  64, 64, LD64, tid, 128);
    cvt_cp2d(aW1,   sA1,  64, 128, LD128, tid, 128);
    cp_f(gv_b1, sb1, 64, tid, 128);
    cp_f(gv_b2, sb2, 64, tid, 128);
    cp_f(ab1,   sab, 64, tid, 128);
    cp_f(aW2,   sa2, 64, tid, 128);
    cp_f(w1_b,  sbw, 64, tid, 128);
    if (tid == 0) sRd[1] = ab2[0];
    {   // gather: 64 rows (P=50 real) x 128 cols
        const int r = tid >> 1, c0 = (tid & 1) * 64;
        int ii = 0, ri = 0; float m = 0.0f;
        if (r < P_) {
            ii = uip[(b * P_ + r) * 2 + 0];
            ri = uip[(b * P_ + r) * 2 + 1];
            m  = (ii > 0) ? 1.0f : 0.0f;
        }
        const float* src = (c0 < 64) ? (itemT + (size_t)ii * 64) : (rateT + (size_t)ri * 64);
        const float4 z = make_float4(0.f, 0.f, 0.f, 0.f);
#pragma unroll
        for (int j = 0; j < 16; j++) {
            const float4 v = (r < P_) ? *(const float4*)(src + 4 * j) : z;
            st4bf(sBA + r * LD128 + c0 + 4 * j, v);
        }
        if ((tid & 1) == 0) sMk[r] = m;
    }
    if (tid < 64) {
        const int uu = uids[b];
        const float* up = userT + (size_t)uu * 64;
        float s = 0.0f;
        for (int k = 0; k < 64; k++) s += up[k] * (float)sW1[tid * LD64 + k];
        sUw[tid] = s;
    }
    __syncthreads();
    gemm_bf<1, 128, 4, 4>(sBA, LD128, sGV1, LD128, sb1, sBB, LD64, tid);
    __syncthreads();
    gemm_bf<0, 64, 4, 4>(sBB, LD64, sGV2, LD64, sb2, sBC, LD64, tid);
    __syncthreads();
    gemm_bf<0, 64, 4, 4>(sBC, LD64, sW1, LD64, sbw, sBA, LD128, tid);
    {
        const int r = tid >> 1, c0 = (tid & 1) * 32;
        for (int c = c0; c < c0 + 32; c++)
            sBA[r * LD128 + 64 + c] = (__bf16)(sMk[r] * sUw[c] + sbw[c]);
    }
    __syncthreads();
    gemm_bf<1, 128, 4, 4>(sBA, LD128, sA1, LD128, sab, sBB, LD64, tid);
    __syncthreads();
    if (tid < 64) {
        float s = 0.0f;
        for (int k = 0; k < 64; k++) s += (float)sBB[tid * LD64 + k] * sa2[k];
        sAl[tid] = __expf(s + sRd[1]) * sMk[tid];
    }
    __syncthreads();
    if (tid == 0) { float s = 0.0f; for (int p = 0; p < 64; p++) s += sAl[p]; sRd[0] = 1.0f / (s + 1e-10f); }
    __syncthreads();
    if (tid < 64) {
        float s = 0.0f;
        for (int p = 0; p < P_; p++) s += sAl[p] * (float)sBC[p * LD64 + tid];
        t_i[(size_t)b * 64 + tid] = s * sRd[0];
    }
}

// ---------------- batched relu(X @ W^T + b) over 15360 rows (h_oI) ----------------
__global__ void __launch_bounds__(128)
k_agg_rows(const float* __restrict__ X, const float* __restrict__ W,
           const float* __restrict__ bias, float* __restrict__ Y) {
    extern __shared__ __align__(16) char smem[];
    __bf16* sW = (__bf16*)smem;        // 64 x LD64
    __bf16* sX = sW + 64 * LD64;       // 64 x LD64
    float*  sb = (float*)(sX + 64 * LD64);
    const int tid  = threadIdx.x;
    const int row0 = blockIdx.x * 64;
    cvt_cp2d(W, sW, 64, 64, LD64, tid, 128);
    cp_f(bias, sb, 64, tid, 128);
    {
        const int r = tid >> 1, c0 = (tid & 1) * 32;
#pragma unroll
        for (int j = 0; j < 8; j++)
            st4bf(sX + r * LD64 + c0 + 4 * j,
                  *(const float4*)(X + (size_t)(row0 + r) * 64 + c0 + 4 * j));
    }
    __syncthreads();
    const int wave = tid >> 5, lane = tid & 31;
    const int mrow = lane & 15, ahi = (lane & 16) ? 8 : 0, bhi = (lane & 16) ? 16 : 0;
    for (int j = wave; j < 16; j += 4) {
        const int mt = j >> 2, nt = j & 3;
        v8f acc = {};
#pragma unroll
        for (int kc = 0; kc < 64; kc += 32) {
            union { v16bf v; v8bf h[2]; } ua, ub;
            const __bf16* ap = sX + (mt * 16 + mrow) * LD64 + kc + ahi;
            ua.h[0] = *(const v8bf*)ap;
            ua.h[1] = *(const v8bf*)(ap + 16);
            const __bf16* bp = sW + (nt * 16 + mrow) * LD64 + kc + ahi;
            ub.h[0] = *(const v8bf*)bp;
            ub.h[1] = *(const v8bf*)(bp + 16);
            acc = __builtin_amdgcn_wmma_f32_16x16x32_bf16(false, ua.v, false, ub.v,
                                                          (short)0, acc, false, false);
        }
        const int n = nt * 16 + mrow, mb = mt * 16 + ahi;
        const float bv = sb[n];
#pragma unroll
        for (int r = 0; r < 8; r++)
            Y[(size_t)(row0 + mb + r) * 64 + n] = fmaxf(acc[r] + bv, 0.0f);
    }
}

// ---------------- beta attention over Q + final combine ----------------
__global__ void __launch_bounds__(128)
k_finish(const int* __restrict__ uup, const float* __restrict__ userT,
         const float* __restrict__ hoI, const float* __restrict__ t_i,
         const float* w5_W, const float* w5_b,
         const float* aW1, const float* ab1, const float* aW2, const float* ab2,
         const float* ag2_W, const float* ag2_b,
         const float* ag1_W, const float* ag1_b,
         const float* c1_W, const float* c1_b, const float* c2_W, const float* c2_b,
         float* __restrict__ out) {
    extern __shared__ __align__(16) char smem[];
    __bf16* sW5 = (__bf16*)smem;         // 64 x LD64
    __bf16* sA1 = sW5 + 64 * LD64;       // 64 x LD128 (attS2_W1)
    __bf16* sG2 = sA1 + 64 * LD128;      // 64 x LD64  (aggS2)
    __bf16* sG1 = sG2 + 64 * LD64;       // 64 x LD64  (agg1)
    __bf16* sC1 = sG1 + 64 * LD64;       // 128 x LD128 (comb_W1)
    __bf16* sC2 = sC1 + 128 * LD128;     // 64 x LD128  (comb_W2)
    __bf16* sBA = sC2 + 64 * LD128;      // 32 x LD128
    __bf16* sBB = sBA + 32 * LD128;      // 32 x LD64
    __bf16* sHb = sBB + 32 * LD64;       // 32 x LD64 (h_oI bf16)
    __bf16* sU2 = sHb + 32 * LD64;       // 32 x LD64 (user emb bf16)
    float*  sHf = (float*)(sU2 + 32 * LD64);  // 32x64 (h_oI f32)
    float*  sb5  = sHf + 32 * 64;
    float*  sab  = sb5 + 64;
    float*  sa2  = sab + 64;
    float*  sg1b = sa2 + 64;
    float*  sg2b = sg1b + 64;
    float*  sc1b = sg2b + 64;   // 128
    float*  sc2b = sc1b + 128;  // 64
    float*  sTi  = sc2b + 64;   // 64
    float*  sHs  = sTi + 64;    // 64
    float*  sCat = sHs + 64;    // 128
    float*  sH1  = sCat + 128;  // 128
    float*  sAl  = sH1 + 128;   // 32
    float*  sMk  = sAl + 32;    // 32
    float*  sRd  = sMk + 32;    // 4

    const int tid = threadIdx.x, b = blockIdx.x;
    cvt_cp2d(w5_W, sW5, 64, 64, LD64, tid, 128);
    cvt_cp2d(aW1,  sA1, 64, 128, LD128, tid, 128);
    cvt_cp2d(ag2_W, sG2, 64, 64, LD64, tid, 128);
    cvt_cp2d(ag1_W, sG1, 64, 64, LD64, tid, 128);
    cvt_cp2d(c1_W, sC1, 128, 128, LD128, tid, 128);
    cvt_cp2d(c2_W, sC2, 64, 128, LD128, tid, 128);
    cp_f(w5_b, sb5, 64, tid, 128);
    cp_f(ab1,  sab, 64, tid, 128);
    cp_f(aW2,  sa2, 64, tid, 128);
    cp_f(ag1_b, sg1b, 64, tid, 128);
    cp_f(ag2_b, sg2b, 64, tid, 128);
    cp_f(c1_b, sc1b, 128, tid, 128);
    cp_f(c2_b, sc2b, 64, tid, 128);
    if (tid == 0) sRd[1] = ab2[0];
    if (tid < 64) sTi[tid] = t_i[(size_t)b * 64 + tid];
    {   // load h_oI rows + social-user rows + mask
        const int r = tid >> 2, c0 = (tid & 3) * 16;
        int uu = 0; float m = 0.0f;
        if (r < Q_) { uu = uup[b * Q_ + r]; m = (uu > 0) ? 1.0f : 0.0f; }
        const float4 z = make_float4(0.f, 0.f, 0.f, 0.f);
#pragma unroll
        for (int j = 0; j < 4; j++) {
            const int c = c0 + 4 * j;
            float4 hv = z, uv = z;
            if (r < Q_) {
                hv = *(const float4*)(hoI + (size_t)(b * Q_ + r) * 64 + c);
                uv = *(const float4*)(userT + (size_t)uu * 64 + c);
            }
            sHf[r * 64 + c + 0] = hv.x; sHf[r * 64 + c + 1] = hv.y;
            sHf[r * 64 + c + 2] = hv.z; sHf[r * 64 + c + 3] = hv.w;
            st4bf(sHb + r * LD64 + c, hv);
            st4bf(sU2 + r * LD64 + c, uv);
        }
        if ((tid & 3) == 0) sMk[r] = m;
    }
    __syncthreads();
    gemm_bf<0, 64, 2, 4>(sHb, LD64, sW5, LD64, sb5, sBA, LD128, tid);       // lin(h_oI,w5) -> left
    gemm_bf<0, 64, 2, 4>(sU2, LD64, sW5, LD64, sb5, sBA + 64, LD128, tid);  // lin(u,w5) -> right
    __syncthreads();
    gemm_bf<1, 128, 2, 4>(sBA, LD128, sA1, LD128, sab, sBB, LD64, tid);     // attS2 hidden
    __syncthreads();
    if (tid < 32) {
        float s = 0.0f;
        for (int k = 0; k < 64; k++) s += (float)sBB[tid * LD64 + k] * sa2[k];
        sAl[tid] = __expf(s + sRd[1]) * sMk[tid];
    }
    __syncthreads();
    if (tid == 0) { float s = 0.0f; for (int p = 0; p < 32; p++) s += sAl[p]; sRd[0] = 1.0f / (s + 1e-10f); }
    __syncthreads();
    if (tid < 64) {   // h_sum = sum_q beta_q * h_oI[q]
        float s = 0.0f;
        for (int q = 0; q < Q_; q++) s += sAl[q] * sHf[q * 64 + tid];
        sHs[tid] = s * sRd[0];
    }
    __syncthreads();
    if (tid < 64) {   // h_iI = relu(agg1 @ t_i + b)
        float s = 0.0f;
        for (int k = 0; k < 64; k++) s += (float)sG1[tid * LD64 + k] * sTi[k];
        sCat[tid] = fmaxf(s + sg1b[tid], 0.0f);
    } else {          // h_iS = relu(aggS2 @ h_sum + b)
        const int n = tid - 64;
        float s = 0.0f;
        for (int k = 0; k < 64; k++) s += (float)sG2[n * LD64 + k] * sHs[k];
        sCat[64 + n] = fmaxf(s + sg2b[n], 0.0f);
    }
    __syncthreads();
    {   // h1 = relu(comb_W1 @ [h_iI,h_iS] + b1)
        float s = 0.0f;
        for (int k = 0; k < 128; k++) s += (float)sC1[tid * LD128 + k] * sCat[k];
        sH1[tid] = fmaxf(s + sc1b[tid], 0.0f);
    }
    __syncthreads();
    if (tid < 64) {   // out = relu(comb_W2 @ h1 + b2)
        float s = 0.0f;
        for (int k = 0; k < 128; k++) s += (float)sC2[tid * LD128 + k] * sH1[k];
        out[(size_t)b * 64 + tid] = fmaxf(s + sc2b[tid], 0.0f);
    }
}

extern "C" void kernel_launch(void* const* d_in, const int* in_sizes, int n_in,
                              void* d_out, int out_size, void* d_ws, size_t ws_size,
                              hipStream_t stream) {
    (void)in_sizes; (void)n_in; (void)out_size; (void)ws_size;
    const int*   uids  = (const int*)d_in[0];
    const int*   uip   = (const int*)d_in[1];
    const int*   uup   = (const int*)d_in[2];
    const int*   uuip  = (const int*)d_in[3];
    const float* userT = (const float*)d_in[4];
    const float* itemT = (const float*)d_in[5];
    const float* rateT = (const float*)d_in[6];
    const float* w1_W = (const float*)d_in[7],  *w1_b = (const float*)d_in[8];
    const float* w4_W = (const float*)d_in[9],  *w4_b = (const float*)d_in[10];
    const float* w5_W = (const float*)d_in[11], *w5_b = (const float*)d_in[12];
    const float* gv_W1 = (const float*)d_in[13], *gv_b1 = (const float*)d_in[14];
    const float* gv_W2 = (const float*)d_in[15], *gv_b2 = (const float*)d_in[16];
    const float* att1_W1 = (const float*)d_in[17], *att1_b1 = (const float*)d_in[18];
    const float* att1_W2 = (const float*)d_in[19], *att1_b2 = (const float*)d_in[20];
    const float* agg1_W = (const float*)d_in[21], *agg1_b = (const float*)d_in[22];
    const float* attS1_W1 = (const float*)d_in[23], *attS1_b1 = (const float*)d_in[24];
    const float* attS1_W2 = (const float*)d_in[25], *attS1_b2 = (const float*)d_in[26];
    const float* aggS1_W = (const float*)d_in[27], *aggS1_b = (const float*)d_in[28];
    const float* attS2_W1 = (const float*)d_in[29], *attS2_b1 = (const float*)d_in[30];
    const float* attS2_W2 = (const float*)d_in[31], *attS2_b2 = (const float*)d_in[32];
    const float* aggS2_W = (const float*)d_in[33], *aggS2_b = (const float*)d_in[34];
    const float* comb_W1 = (const float*)d_in[35], *comb_b1 = (const float*)d_in[36];
    const float* comb_W2 = (const float*)d_in[37], *comb_b2 = (const float*)d_in[38];

    float* t_s = (float*)d_ws;                 // B*Q*64
    float* t_i = t_s + (size_t)B_ * Q_ * 64;   // B*64
    float* hoI = t_i + (size_t)B_ * 64;        // B*Q*64

    constexpr size_t SOC_SMEM =
        (size_t)(64 * LD128 + 64 * LD64 + 64 * LD64 + 64 * LD128 +
                 32 * LD128 + 32 * LD64 + 32 * LD64) * 2 +
        (size_t)(64 * 6 + 32 + 32 + 4) * 4 + 64;
    constexpr size_t ITEM_SMEM =
        (size_t)(64 * LD128 + 64 * LD64 + 64 * LD64 + 64 * LD128 +
                 64 * LD128 + 64 * LD64 + 64 * LD64) * 2 +
        (size_t)(64 * 6 + 64 + 64 + 4) * 4 + 64;
    constexpr size_t K3_SMEM = (size_t)(64 * LD64 + 64 * LD64) * 2 + 64 * 4 + 64;
    constexpr size_t K4_SMEM =
        (size_t)(64 * LD64 + 64 * LD128 + 64 * LD64 + 64 * LD64 + 128 * LD128 + 64 * LD128 +
                 32 * LD128 + 32 * LD64 + 32 * LD64 + 32 * LD64) * 2 +
        (size_t)(32 * 64 + 64 * 9 + 128 * 3 + 32 + 32 + 4) * 4 + 64;

    k_social<<<B_, 128, SOC_SMEM, stream>>>(uup, uuip, userT, itemT, rateT,
                                            gv_W1, gv_b1, gv_W2, gv_b2, w4_W, w4_b,
                                            attS1_W1, attS1_b1, attS1_W2, attS1_b2, t_s);
    k_item<<<B_, 128, ITEM_SMEM, stream>>>(uids, uip, userT, itemT, rateT,
                                           gv_W1, gv_b1, gv_W2, gv_b2, w1_W, w1_b,
                                           att1_W1, att1_b1, att1_W2, att1_b2, t_i);
    k_agg_rows<<<(B_ * Q_) / 64, 128, K3_SMEM, stream>>>(t_s, aggS1_W, aggS1_b, hoI);
    k_finish<<<B_, 128, K4_SMEM, stream>>>(uup, userT, hoI, t_i,
                                           w5_W, w5_b,
                                           attS2_W1, attS2_b1, attS2_W2, attS2_b2,
                                           aggS2_W, aggS2_b, agg1_W, agg1_b,
                                           comb_W1, comb_b1, comb_W2, comb_b2,
                                           (float*)d_out);
}
